// GoogleAttention_154618823219
// MI455X (gfx1250) — matrix operand order repaired and checked by reading, not compiled, not run on hardware
//
#include <hip/hip_runtime.h>

typedef __bf16 bf16_t;
typedef __attribute__((ext_vector_type(16))) __bf16 v16bf;
typedef __attribute__((ext_vector_type(8)))  __bf16 v8bf;
typedef __attribute__((ext_vector_type(8)))  float  v8f;

#define DEVINL __device__ __forceinline__

// ---------------------------------------------------------------- constants
// B=4, H=W=D=16 -> N=4096, C=512, Ck=C/8=64, Cv=C/2=256
constexpr int NB  = 4;
constexpr int NN  = 4096;
constexpr int CC  = 512;
constexpr int CKD = 64;     // key/query dim
constexpr int CVD = 256;    // value dim
constexpr int BN  = NB * NN;         // 16384 rows
constexpr int CAT = CKD + CKD + CVD; // 384 fused proj cols

constexpr float LOG2E = 1.44269504088896340736f;

// workspace layout (bytes, all 256-aligned)
constexpr size_t OFF_WCAT = 0;                                    // bf16 [384][512]
constexpr size_t OFF_WOT  = OFF_WCAT + (size_t)CAT*CC*2;          // bf16 [512][256]
constexpr size_t OFF_BCAT = OFF_WOT  + (size_t)CC*CVD*2;          // f32  [384]
constexpr size_t OFF_Q    = OFF_BCAT + 1536;                      // bf16 [16384][64]  (g*log2e)
constexpr size_t OFF_K    = OFF_Q    + (size_t)BN*CKD*2;          // bf16 [16384][64]  (f)
constexpr size_t OFF_VT   = OFF_K    + (size_t)BN*CKD*2;          // bf16 [4][256][4096]
constexpr size_t OFF_O    = OFF_VT   + (size_t)NB*CVD*NN*2;       // bf16 [16384][256]
constexpr size_t OFF_XBF  = OFF_O    + (size_t)BN*CVD*2;          // bf16 [16384][512]

// ---------------------------------------------------------------- helpers
DEVINL bf16_t f2bf(float f) {
  union { float f; unsigned u; } v; v.f = f;
  unsigned r = v.u + 0x7FFFu + ((v.u >> 16) & 1u);   // round-to-nearest-even
  unsigned short h = (unsigned short)(r >> 16);
  bf16_t out; __builtin_memcpy(&out, &h, 2);
  return out;
}

DEVINL v8f wmma_bf16(v16bf a, v16bf b, v8f c) {
  // D = A(16x32) * B(32x16) + C, f32 accumulate
  return __builtin_amdgcn_wmma_f32_16x16x32_bf16(false, a, false, b, (short)0, c,
                                                 false, false);
}

// A fragment (16x32 bf16): lane 0-15 row=lane K:0..15, lane 16-31 K:16..31
DEVINL v16bf load_a_frag(const bf16_t* base, int row_stride, int kbase) {
  int lane = threadIdx.x & 31;
  int row = lane & 15, kh = lane >> 4;
  const bf16_t* p = base + row * row_stride + kbase + kh * 16;
  return *(const v16bf*)(p);          // K = kh*16 .. +15
}

// B fragment (32x16 bf16) from K-major storage [col][k]:
// lane 0-15 col=lane K=0..15, lane 16-31 col=lane-16 K=16..31 (contiguous 32B)
DEVINL v16bf load_b_frag(const bf16_t* base, int col_stride, int kbase) {
  int lane = threadIdx.x & 31;
  int col = lane & 15, ks = lane >> 4;
  return *(const v16bf*)(base + col * col_stride + kbase + ks * 16);
}

// ---------------------------------------------------------------- kernel 0a: weight repack
__global__ __launch_bounds__(256) void repack_kernel(
    const float* __restrict__ Wf, const float* __restrict__ bfv,
    const float* __restrict__ Wg, const float* __restrict__ bgv,
    const float* __restrict__ Wh, const float* __restrict__ bhv,
    const float* __restrict__ Wo,
    bf16_t* __restrict__ WcatT, float* __restrict__ bcat, bf16_t* __restrict__ WoT) {
  constexpr int S0 = CAT * CC;        // 196608 WcatT elems
  constexpr int S1 = S0 + CC * CVD;   // + 131072 WoT elems
  constexpr int S2 = S1 + CAT;        // + 384 bias elems
  int i = blockIdx.x * 256 + threadIdx.x;
  if (i < S0) {
    int c = i / CC, k = i % CC;
    // queries pre-scaled by log2(e) so softmax can use raw v_exp_f32 (exp2)
    float v = (c < 64) ? Wf[k * 64 + c]
            : (c < 128) ? Wg[k * 64 + (c - 64)] * LOG2E
                        : Wh[k * 256 + (c - 128)];
    WcatT[i] = f2bf(v);
  } else if (i < S1) {
    int j = i - S0;
    int n = j / CVD, k = j % CVD;     // WoT[n][k] = Wo[k][n]
    WoT[j] = f2bf(Wo[k * CC + n]);
  } else if (i < S2) {
    int c = i - S1;
    bcat[c] = (c < 64) ? bfv[c]
            : (c < 128) ? bgv[c - 64] * LOG2E
                        : bhv[c - 128];
  }
}

// ---------------------------------------------------------------- kernel 0b: x -> bf16 (once)
__global__ __launch_bounds__(256) void cvtx_kernel(const float* __restrict__ x,
                                                   bf16_t* __restrict__ xbf) {
  size_t i = ((size_t)blockIdx.x * 256 + threadIdx.x) * 8;
  float fv[8];
  *(float4*)(fv + 0) = ((const float4*)(x + i))[0];
  *(float4*)(fv + 4) = ((const float4*)(x + i))[1];
  alignas(16) bf16_t tmp[8];
#pragma unroll
  for (int k = 0; k < 8; ++k) tmp[k] = f2bf(fv[k]);
  *(v8bf*)(xbf + i) = *(const v8bf*)tmp;
}

// ---------------------------------------------------------------- kernel 1: fused QKV projection
// P[16384,384] = xbf[16384,512] @ Wcat ; scatter -> q,k (row major) and vT (K-major)
__global__ __launch_bounds__(256) void proj_kernel(
    const bf16_t* __restrict__ xbf, const bf16_t* __restrict__ WcatT,
    const float* __restrict__ bcat,
    bf16_t* __restrict__ qbuf, bf16_t* __restrict__ kbuf, bf16_t* __restrict__ vT) {
  __shared__ bf16_t a_lds[128 * 32];   // [row][k]
  __shared__ bf16_t b_lds[64 * 32];    // [col][k]

  const int r0 = blockIdx.x * 128;
  const int c0 = blockIdx.y * 64;
  const int t = threadIdx.x, wave = t >> 5, lane = t & 31;
  const int half = lane >> 4, ln = lane & 15;

  v8f acc[4];
#pragma unroll
  for (int s = 0; s < 4; ++s)
#pragma unroll
    for (int j = 0; j < 8; ++j) acc[s][j] = 0.f;

  for (int k0 = 0; k0 < CC; k0 += 32) {
    __syncthreads();
    {   // A tile: 128x32 bf16 straight copy
      int row = t >> 1, hh = (t & 1) * 16;
      *(v16bf*)(a_lds + row * 32 + hh) =
          *(const v16bf*)(xbf + (size_t)(r0 + row) * CC + k0 + hh);
    }
    {   // B tile: 64x32 bf16 K-major
      int col = t >> 2, q = (t & 3) * 8;
      *(v8bf*)(b_lds + col * 32 + q) =
          *(const v8bf*)(WcatT + (size_t)(c0 + col) * CC + k0 + q);
    }
    __syncthreads();
    v16bf a = load_a_frag(a_lds + wave * 16 * 32, 32, 0);
#pragma unroll
    for (int s = 0; s < 4; ++s) {
      v16bf b = load_b_frag(b_lds + s * 16 * 32, 32, 0);
      acc[s] = wmma_bf16(a, b, acc[s]);
    }
  }

#pragma unroll
  for (int s = 0; s < 4; ++s) {
    int c = c0 + s * 16 + ln;
    float bias = bcat[c];
#pragma unroll
    for (int j = 0; j < 8; ++j) {
      int r = r0 + wave * 16 + half * 8 + j;
      bf16_t bv = f2bf(acc[s][j] + bias);
      if (c < 64) {                 // f = keys
        kbuf[(size_t)r * 64 + c] = bv;
      } else if (c < 128) {         // g = queries (already *log2e)
        qbuf[(size_t)r * 64 + (c - 64)] = bv;
      } else {                      // hv = values, stored K-major per batch
        int b = r >> 12, n = r & 4095;
        vT[((size_t)b * CVD + (c - 128)) * NN + n] = bv;
      }
    }
  }
}

// ---------------------------------------------------------------- kernel 2: flash attention
// per block: one batch, 64 query rows; 8 waves.
// Waves 0-3 own one 16-row tile each: compute S + online softmax, publish P/scale via LDS.
// All 8 waves (rowt = wave&3, colh = wave>>2) accumulate O over a 128-col value half.
__global__ __launch_bounds__(256) void attn_kernel(
    const bf16_t* __restrict__ qbuf, const bf16_t* __restrict__ kbuf,
    const bf16_t* __restrict__ vT, bf16_t* __restrict__ obuf) {
  __shared__ bf16_t q_lds[64 * 64];       // [qrow][k]        8 KB
  __shared__ bf16_t k_lds[64 * 64];       // [key][k]         8 KB
  __shared__ bf16_t v_lds[256 * 64];      // [vcol][key]     32 KB
  __shared__ bf16_t p_lds[4 * 16 * 64];   // [rowtile][row][key] 8 KB
  __shared__ float  stat_lds[64];         // per-row scale (iter) / final sum

  const int b  = blockIdx.x >> 6;
  const int q0 = (blockIdx.x & 63) * 64;
  const int t = threadIdx.x, wave = t >> 5, lane = t & 31;
  const int rowt = wave & 3, colh = wave >> 2;
  const int half = lane >> 4, ln = lane & 15;
  const size_t qrow0 = (size_t)b * NN + q0;

  {   // stage the 64x64 query tile once
    int row = t >> 2, ch = (t & 3) * 16;
    *(v16bf*)(q_lds + row * 64 + ch) =
        *(const v16bf*)(qbuf + (qrow0 + row) * 64 + ch);
  }

  v8f acc[8];
#pragma unroll
  for (int f = 0; f < 8; ++f)
#pragma unroll
    for (int j = 0; j < 8; ++j) acc[f][j] = 0.f;
  float mrun[8], lrun[8];
#pragma unroll
  for (int j = 0; j < 8; ++j) { mrun[j] = -1e30f; lrun[j] = 0.f; }

  bf16_t* pw = p_lds + rowt * 16 * 64;

  for (int kb = 0; kb < NN; kb += 64) {
    __syncthreads();                                // (1) prior reads done
    {   // keys 64x64 bf16
      int row = t >> 2, ch = (t & 3) * 16;
      *(v16bf*)(k_lds + row * 64 + ch) =
          *(const v16bf*)(kbuf + ((size_t)b * NN + kb + row) * 64 + ch);
    }
    {   // values 256x64 bf16 (already K-major in global)
      const bf16_t* src = vT + ((size_t)b * CVD + t) * NN + kb;
      bf16_t* dst = v_lds + t * 64;
#pragma unroll
      for (int i = 0; i < 4; ++i)
        *(v16bf*)(dst + i * 16) = *(const v16bf*)(src + i * 16);
    }
    __syncthreads();                                // (2) tiles visible

    if (colh == 0) {
      // ---- scores S = Q(16x64) @ K^T(64x64): 4 col subtiles x 2 K steps
      v8f s[4];
#pragma unroll
      for (int sub = 0; sub < 4; ++sub)
#pragma unroll
        for (int j = 0; j < 8; ++j) s[sub][j] = 0.f;
#pragma unroll
      for (int ks = 0; ks < 2; ++ks) {
        v16bf a = load_a_frag(q_lds + rowt * 16 * 64, 64, ks * 32);
#pragma unroll
        for (int sub = 0; sub < 4; ++sub) {
          v16bf bm = load_b_frag(k_lds + sub * 16 * 64, 64, ks * 32);
          s[sub] = wmma_bf16(a, bm, s[sub]);
        }
      }
      // ---- online softmax in exp2 domain (queries pre-scaled by log2e)
#pragma unroll
      for (int j = 0; j < 8; ++j) {
        float r = fmaxf(fmaxf(s[0][j], s[1][j]), fmaxf(s[2][j], s[3][j]));
#pragma unroll
        for (int m = 1; m < 16; m <<= 1) r = fmaxf(r, __shfl_xor(r, m, 32));
        float mnew = fmaxf(mrun[j], r);
        float sc = exp2f(mrun[j] - mnew);
        mrun[j] = mnew;
        lrun[j] *= sc;
        if (ln == 0) stat_lds[rowt * 16 + half * 8 + j] = sc;
        float rs = 0.f;
#pragma unroll
        for (int sub = 0; sub < 4; ++sub) {
          float p = exp2f(s[sub][j] - mnew);
          rs += p;
          pw[(half * 8 + j) * 64 + sub * 16 + ln] = f2bf(p);
        }
#pragma unroll
        for (int m = 1; m < 16; m <<= 1) rs += __shfl_xor(rs, m, 32);
        lrun[j] += rs;
      }
    }
    __syncthreads();                                // (3) P + scale published

    // ---- rescale accumulators, then O += P(16x64) @ V(64x128)
    float sc[8];
#pragma unroll
    for (int j = 0; j < 8; ++j) sc[j] = stat_lds[rowt * 16 + half * 8 + j];
#pragma unroll
    for (int f = 0; f < 8; ++f)
#pragma unroll
      for (int j = 0; j < 8; ++j) acc[f][j] *= sc[j];
#pragma unroll
    for (int ks = 0; ks < 2; ++ks) {
      v16bf a = load_a_frag(pw, 64, ks * 32);
#pragma unroll
      for (int sub = 0; sub < 8; ++sub) {
        int col = colh * 128 + sub * 16;
        v16bf bm = load_b_frag(v_lds + col * 64, 64, ks * 32);
        acc[sub] = wmma_bf16(a, bm, acc[sub]);
      }
    }
  }

  // publish final row sums, normalize, store o (bf16, row major)
  __syncthreads();
  if (colh == 0 && ln == 0) {
#pragma unroll
    for (int j = 0; j < 8; ++j) stat_lds[rowt * 16 + half * 8 + j] = lrun[j];
  }
  __syncthreads();
  float inv[8];
#pragma unroll
  for (int j = 0; j < 8; ++j) inv[j] = 1.0f / stat_lds[rowt * 16 + half * 8 + j];
#pragma unroll
  for (int sub = 0; sub < 8; ++sub) {
    int c = colh * 128 + sub * 16 + ln;
#pragma unroll
    for (int j = 0; j < 8; ++j) {
      int r = q0 + rowt * 16 + half * 8 + j;
      obuf[((size_t)b * NN + r) * CVD + c] = f2bf(acc[sub][j] * inv[j]);
    }
  }
}

// ---------------------------------------------------------------- kernel 3: output proj + residual
__global__ __launch_bounds__(256) void outproj_kernel(
    const bf16_t* __restrict__ obuf, const bf16_t* __restrict__ WoT,
    const float* __restrict__ bo, const float* __restrict__ gamma,
    const float* __restrict__ x, float* __restrict__ out) {
  __shared__ bf16_t a_lds[128 * 32];
  __shared__ bf16_t b_lds[64 * 32];

  const int r0 = blockIdx.x * 128;
  const int c0 = blockIdx.y * 64;
  const int t = threadIdx.x, wave = t >> 5, lane = t & 31;
  const int half = lane >> 4, ln = lane & 15;

  v8f acc[4];
#pragma unroll
  for (int s = 0; s < 4; ++s)
#pragma unroll
    for (int j = 0; j < 8; ++j) acc[s][j] = 0.f;

  for (int k0 = 0; k0 < CVD; k0 += 32) {
    __syncthreads();
    {
      int row = t >> 1, hh = (t & 1) * 16;
      *(v16bf*)(a_lds + row * 32 + hh) =
          *(const v16bf*)(obuf + (size_t)(r0 + row) * CVD + k0 + hh);
    }
    {
      int col = t >> 2, q = (t & 3) * 8;
      *(v8bf*)(b_lds + col * 32 + q) =
          *(const v8bf*)(WoT + (size_t)(c0 + col) * CVD + k0 + q);
    }
    __syncthreads();
    v16bf a = load_a_frag(a_lds + wave * 16 * 32, 32, 0);
#pragma unroll
    for (int s = 0; s < 4; ++s) {
      v16bf b = load_b_frag(b_lds + s * 16 * 32, 32, 0);
      acc[s] = wmma_bf16(a, b, acc[s]);
    }
  }

  const float gam = gamma[0];
#pragma unroll
  for (int s = 0; s < 4; ++s) {
    int c = c0 + s * 16 + ln;
    float bias = bo[c];
#pragma unroll
    for (int j = 0; j < 8; ++j) {
      int r = r0 + wave * 16 + half * 8 + j;
      size_t idx = (size_t)r * CC + c;
      out[idx] = gam * (acc[s][j] + bias) + x[idx];
    }
  }
}

// ---------------------------------------------------------------- launcher
extern "C" void kernel_launch(void* const* d_in, const int* in_sizes, int n_in,
                              void* d_out, int out_size, void* d_ws, size_t ws_size,
                              hipStream_t stream) {
  const float* x     = (const float*)d_in[0];
  const float* Wf    = (const float*)d_in[1];
  const float* bfv   = (const float*)d_in[2];
  const float* Wg    = (const float*)d_in[3];
  const float* bgv   = (const float*)d_in[4];
  const float* Wh    = (const float*)d_in[5];
  const float* bhv   = (const float*)d_in[6];
  const float* Wo    = (const float*)d_in[7];
  const float* bo    = (const float*)d_in[8];
  const float* gamma = (const float*)d_in[9];
  float* out = (float*)d_out;

  char* ws = (char*)d_ws;
  bf16_t* WcatT = (bf16_t*)(ws + OFF_WCAT);
  bf16_t* WoT   = (bf16_t*)(ws + OFF_WOT);
  float*  bcat  = (float*) (ws + OFF_BCAT);
  bf16_t* qbuf  = (bf16_t*)(ws + OFF_Q);
  bf16_t* kbuf  = (bf16_t*)(ws + OFF_K);
  bf16_t* vT    = (bf16_t*)(ws + OFF_VT);
  bf16_t* obuf  = (bf16_t*)(ws + OFF_O);
  bf16_t* xbf   = (bf16_t*)(ws + OFF_XBF);

  // 0a) repack weights to bf16 K-major (+ fold log2e into queries)
  constexpr int repack_total = CAT * CC + CC * CVD + CAT;
  repack_kernel<<<dim3((repack_total + 255) / 256), dim3(256), 0, stream>>>(
      Wf, bfv, Wg, bgv, Wh, bhv, Wo, WcatT, bcat, WoT);

  // 0b) convert x to bf16 once
  cvtx_kernel<<<dim3(BN * CC / 8 / 256), dim3(256), 0, stream>>>(x, xbf);

  // 1) fused QKV projection: 16384x384 = x @ [Wf|Wg|Wh]
  proj_kernel<<<dim3(BN / 128, CAT / 64), dim3(256), 0, stream>>>(
      xbf, WcatT, bcat, qbuf, kbuf, vT);

  // 2) flash attention: 4 batches x 64 query tiles
  attn_kernel<<<dim3(NB * (NN / 64)), dim3(256), 0, stream>>>(
      qbuf, kbuf, vT, obuf);

  // 3) output projection + gamma*o + x
  outproj_kernel<<<dim3(BN / 128, CC / 64), dim3(256), 0, stream>>>(
      obuf, WoT, bo, gamma, x, out);
}
